// VectorQuantize_22634477650355
// MI455X (gfx1250) — matrix operand order repaired and checked by reading, not compiled, not run on hardware
//
#include <hip/hip_runtime.h>
#include <hip/hip_bf16.h>

// Problem constants (from the reference)
#define DIM   64
#define NE    8192
#define TLEN  8192
#define BATCH 8
#define NROWS (BATCH * TLEN)          // 65536 flattened vectors
#define QSIZE (BATCH * DIM * TLEN)    // 4194304 quantize elements
#define IND_OFF  QSIZE
#define LOSS_OFF (QSIZE + NROWS)
#define PPL_OFF  (LOSS_OFF + 1)

#define ESCALE 8192.0f                // exact power-of-two codebook prescale
#define COLSTR 72                     // padded halves per column in LDS (bank-conflict-free)

typedef __attribute__((ext_vector_type(16))) _Float16 v16h;  // f16 WMMA A/B operand
typedef __attribute__((ext_vector_type(8)))  float    v8f;   // f32 WMMA C/D (16x16)

union V16 { v16h v; float4 f4[2]; };  // 16-byte-granular LDS moves for v16h

// ---------------------------------------------------------------------------
// Kernel 1: scaled per-code norms 8192*||e_j||^2 -> ws; zero bitmask + loss.
// ---------------------------------------------------------------------------
__global__ void vq_norms_kernel(const float* __restrict__ embed,
                                float* __restrict__ norms,
                                unsigned* __restrict__ mask,
                                float* __restrict__ out) {
    const int j = blockIdx.x * 256 + threadIdx.x;
    float s = 0.0f;
#pragma unroll
    for (int d = 0; d < DIM; ++d) {
        float e = embed[(size_t)d * NE + j];
        s += e * e;
    }
    norms[j] = s * ESCALE;            // monotone-scaled: 8192*||e||^2 - 2*(x . 8192e)
    if (blockIdx.x == 0) {
        mask[threadIdx.x] = 0u;
        if (threadIdx.x == 0) out[LOSS_OFF] = 0.0f;
    }
}

// ---------------------------------------------------------------------------
// Kernel 2: split-fp16 GEMM (V_WMMA_F32_16X16X32_F16, fp32-accurate via
// xh*eh + xl*eh + xh*el with fp32 accumulate) fused with argmin + gather +
// loss + code-usage bitmask.
//   Block: 256 threads = 8 waves (wave32). Block owns 128 rows; wave owns 16.
//   A (xh,xl) in registers all kernel; codebook tiles converted to (eh,el)
//   halves in LDS, K-contiguous per column for b128 reads.
// ---------------------------------------------------------------------------
__launch_bounds__(256, 2)
__global__ void vq_main_kernel(const float* __restrict__ x,
                               const float* __restrict__ embed,
                               const float* __restrict__ norms,
                               unsigned* __restrict__ mask,
                               float* __restrict__ out) {
    __shared__ _Float16 ldsBh[128 * COLSTR];   // eh tile: [col][K(0..63)+pad]
    __shared__ _Float16 ldsBl[128 * COLSTR];   // el tile
    __shared__ int      rowIdx[8 * 16];

    const int lane = threadIdx.x & 31;
    const int wave = threadIdx.x >> 5;
    const int l16  = lane & 15;
    const int hi   = lane >> 4;
    const int rowBase = blockIdx.x * 128 + wave * 16;
    const int n = rowBase + l16;               // flat row this lane represents
    const int b = n >> 13;
    const int t = n & (TLEN - 1);
    const float* xrow = x + (size_t)b * (DIM * TLEN) + t;  // x[b,d,t] at xrow[d*TLEN]

    // ---- A operands: split x row into fp16 hi/lo, contiguous K order (same as B):
    //      half h of block kb32 holds K = kb32*32 + hi*16 + h
    v16h a_h[2], a_l[2];
#pragma unroll
    for (int kb32 = 0; kb32 < 2; ++kb32) {
        v16h xh, xl;
#pragma unroll
        for (int h = 0; h < 16; ++h) {
            const int K = kb32 * 32 + hi * 16 + h;
            const float xv = xrow[(size_t)K * TLEN];
            const _Float16 hv = (_Float16)xv;
            xh[h] = hv;
            xl[h] = (_Float16)(xv - (float)hv);
        }
        a_h[kb32] = xh;
        a_l[kb32] = xl;
    }

    float minv[8];
    int   mini[8];
#pragma unroll
    for (int r = 0; r < 8; ++r) { minv[r] = 3.4e38f; mini[r] = 0; }

    // Conversion-stage mapping: thread -> (column, K-half) of the 64x128 tile.
    const int ccol   = threadIdx.x & 127;
    const int ckhalf = threadIdx.x >> 7;       // 0: K 0..31, 1: K 32..63

    for (int j0 = 0; j0 < NE; j0 += 128) {
        __syncthreads();
        // ---- Convert embed[:, j0:j0+128] to scaled fp16 hi/lo pairs in LDS.
        //      Dest halves are K-contiguous per column (matches B read layout).
        {
            V16 th[2], tl[2];
#pragma unroll
            for (int k = 0; k < 32; ++k) {
                const int K = ckhalf * 32 + k;
                const float ev = embed[(size_t)K * NE + (j0 + ccol)] * ESCALE;
                const _Float16 hv = (_Float16)ev;
                th[k >> 4].v[k & 15] = hv;
                tl[k >> 4].v[k & 15] = (_Float16)(ev - (float)hv);
            }
            _Float16* dh = ldsBh + ccol * COLSTR + ckhalf * 32;
            _Float16* dl = ldsBl + ccol * COLSTR + ckhalf * 32;
#pragma unroll
            for (int g = 0; g < 2; ++g) {
                ((float4*)(dh + g * 16))[0] = th[g].f4[0];
                ((float4*)(dh + g * 16))[1] = th[g].f4[1];
                ((float4*)(dl + g * 16))[0] = tl[g].f4[0];
                ((float4*)(dl + g * 16))[1] = tl[g].f4[1];
            }
        }
        __syncthreads();

        // Prefetch next code tile (global_prefetch_b8).
        if (j0 + 128 < NE) {
            const int d = threadIdx.x >> 2, line = threadIdx.x & 3;
            __builtin_prefetch(embed + (size_t)d * NE + (j0 + 128) + line * 32, 0, 1);
        }

#pragma unroll 1
        for (int sub = 0; sub < 8; ++sub) {
            const int jc = sub * 16 + l16;
            const float nj = norms[j0 + jc];

            // B operands in contiguous K order (same as A): half h of block kb32 holds
            // K = kb32*32 + hi*16 + h, column = l16 -> contiguous in our LDS.
            const _Float16* pbh = ldsBh + jc * COLSTR + hi * 16;
            const _Float16* pbl = ldsBl + jc * COLSTR + hi * 16;
            V16 bh0, bh1, bl0, bl1;
            bh0.f4[0] = ((const float4*)pbh)[0];        bh0.f4[1] = ((const float4*)pbh)[1];
            bh1.f4[0] = ((const float4*)(pbh + 32))[0]; bh1.f4[1] = ((const float4*)(pbh + 32))[1];
            bl0.f4[0] = ((const float4*)pbl)[0];        bl0.f4[1] = ((const float4*)pbl)[1];
            bl1.f4[0] = ((const float4*)(pbl + 32))[0]; bl1.f4[1] = ((const float4*)(pbl + 32))[1];

            // dot' = xh.eh + xl.eh + xh.el   (fp32 accumulate, K=64 via 2 blocks)
            v8f c = {};
            c = __builtin_amdgcn_wmma_f32_16x16x32_f16(false, a_h[0], false, bh0.v, (short)0, c, false, false);
            c = __builtin_amdgcn_wmma_f32_16x16x32_f16(false, a_h[1], false, bh1.v, (short)0, c, false, false);
            c = __builtin_amdgcn_wmma_f32_16x16x32_f16(false, a_l[0], false, bh0.v, (short)0, c, false, false);
            c = __builtin_amdgcn_wmma_f32_16x16x32_f16(false, a_l[1], false, bh1.v, (short)0, c, false, false);
            c = __builtin_amdgcn_wmma_f32_16x16x32_f16(false, a_h[0], false, bl0.v, (short)0, c, false, false);
            c = __builtin_amdgcn_wmma_f32_16x16x32_f16(false, a_h[1], false, bl1.v, (short)0, c, false, false);

            // Fused argmin on  8192*||e||^2 - 2*dot'  (monotone in true distance).
#pragma unroll
            for (int r = 0; r < 8; ++r) {
                const float v = nj - 2.0f * c[r];
                if (v < minv[r]) { minv[r] = v; mini[r] = j0 + jc; }
            }
        }
    }

    // Cross-lane argmin within 16-lane halves (C layout: lanes0-15 M=r, 16-31 M=r+8).
#pragma unroll
    for (int r = 0; r < 8; ++r) {
        float v = minv[r];
        int   i = mini[r];
#pragma unroll
        for (int m = 8; m >= 1; m >>= 1) {
            const float ov = __shfl_xor(v, m, 32);
            const int   oi = __shfl_xor(i, m, 32);
            if (ov < v || (ov == v && oi < i)) { v = ov; i = oi; }
        }
        minv[r] = v; mini[r] = i;
    }

    if (l16 == 0) {
#pragma unroll
        for (int r = 0; r < 8; ++r) rowIdx[wave * 16 + hi * 8 + r] = mini[r];
    }
    __syncthreads();

    const int myIdx = rowIdx[wave * 16 + l16];

    if (hi == 0) out[IND_OFF + n] = (float)myIdx;

    // Gather quantize + commitment-loss partial; (l16, l16+16) split 64 dims.
    float part = 0.0f;
#pragma unroll
    for (int i = 0; i < 32; ++i) {
        const int d = hi + 2 * i;
        const float q = embed[(size_t)d * NE + myIdx];
        const size_t off = (size_t)b * (DIM * TLEN) + (size_t)d * TLEN + t;
        const float diff = q - x[off];
        part += diff * diff;
        out[off] = q;                 // straight-through forward == quantize
    }
#pragma unroll
    for (int m = 16; m >= 1; m >>= 1) part += __shfl_xor(part, m, 32);
    if (lane == 0) atomicAdd(out + LOSS_OFF, part * (0.02f / (float)QSIZE));

    if (hi == 0) atomicOr(mask + (myIdx >> 5), 1u << (myIdx & 31));
}

// ---------------------------------------------------------------------------
// Kernel 3: perplexity = popcount of usage bitmask.
// ---------------------------------------------------------------------------
__global__ void vq_ppl_kernel(const unsigned* __restrict__ mask,
                              float* __restrict__ out) {
    __shared__ int red[256];
    red[threadIdx.x] = __popc(mask[threadIdx.x]);
    __syncthreads();
    for (int s = 128; s > 0; s >>= 1) {
        if (threadIdx.x < s) red[threadIdx.x] += red[threadIdx.x + s];
        __syncthreads();
    }
    if (threadIdx.x == 0) out[PPL_OFF] = (float)red[0];
}

// ---------------------------------------------------------------------------
extern "C" void kernel_launch(void* const* d_in, const int* in_sizes, int n_in,
                              void* d_out, int out_size, void* d_ws, size_t ws_size,
                              hipStream_t stream) {
    const float* x     = (const float*)d_in[0];   // [8, 64, 8192] f32
    const float* embed = (const float*)d_in[1];   // [64, 8192] f32
    float* out = (float*)d_out;

    unsigned* mask  = (unsigned*)d_ws;            // 256 u32 bitmask
    float*    norms = (float*)d_ws + 256;         // 8192 f32 scaled code norms

    vq_norms_kernel<<<NE / 256, 256, 0, stream>>>(embed, norms, mask, out);
    vq_main_kernel<<<NROWS / 128, 256, 0, stream>>>(x, embed, norms, mask, out);
    vq_ppl_kernel<<<1, 256, 0, stream>>>(mask, out);
}
